// ScaledDotProductAttention_Gauss_6493990552301
// MI455X (gfx1250) — matrix operand order repaired and checked by reading, not compiled, not run on hardware
//
#include <hip/hip_runtime.h>
#include <hip/hip_bf16.h>
#include <hip/hip_fp16.h>
#include <math.h>

typedef __attribute__((ext_vector_type(16))) __bf16 v16bf;
typedef __attribute__((ext_vector_type(8)))  __bf16 v8bf;
typedef __attribute__((ext_vector_type(8)))  float  v8f;

#define BH   8
#define SEQ  1024
#define DH   64
#define KGN  10

#define KP 72   // ldsK row pitch (bf16 elems)   [n][d]
#define VP 40   // ldsVt row pitch (bf16 elems)  [d][k]
#define PP 40   // ldsP  row pitch (bf16 elems)  per-wave [m][k]

// f32 parameter staging layout in LDS
#define PAR_AD 0    // b_D - mu_D
#define PAR_BD 10   // 1/sigma_D
#define PAR_CD 20   // 1/(sqrt(2pi) sigma_D)
#define PAR_AE 30
#define PAR_BE 40
#define PAR_CE 50
#define PAR_W2 60
#define PAR_B2 70
#define PAR_N  71

__device__ __forceinline__ v16bf mk16(v8bf lo, v8bf hi) {
    v16bf r;
#pragma unroll
    for (int i = 0; i < 8; ++i) { r[i] = lo[i]; r[i + 8] = hi[i]; }
    return r;
}

// RBF -> 10x10 MLP (packed fp16, v_pk_fma_f16) -> exact GELU (f32 erf) -> 10x1.
// Processes 8 row-elements as 4 packed pairs; W1/b1 LDS reads amortized over all 8.
// Parameter base pointers are laundered through an opaque VGPR offset so the
// compiler re-reads them from LDS per call instead of hoisting ~135 values
// into long-lived VGPRs across the j-loop (which pushed allocation past 256).
__device__ __forceinline__ void bias8(const float xs[8], float out[8],
                                      const float* par,
                                      const __half2* w1h,
                                      const __half2* b1h, int off) {
    unsigned lz = 0;
    asm volatile("" : "+v"(lz));           // opaque zero: blocks LICM/CSE hoisting
    const float*   parv = par + lz;
    const __half2* w1v  = w1h + lz;
    const __half2* b1v  = b1h + lz;

    __half2 psi[KGN][4];
#pragma unroll
    for (int k = 0; k < KGN; ++k) {
        float a = parv[off + k];            // (b - mu)
        float b = parv[off + 10 + k];       // 1/sigma
        float c = parv[off + 20 + k];       // inv_norm
#pragma unroll
        for (int rr = 0; rr < 4; ++rr) {
            float z0 = (xs[2 * rr]     + a) * b;
            float z1 = (xs[2 * rr + 1] + a) * b;
            float p0 = __expf(-0.5f * z0 * z0) * c;
            float p1 = __expf(-0.5f * z1 * z1) * c;
            psi[k][rr] = __floats2half2_rn(p0, p1);
        }
    }
    float acc[8];
    float b2v = parv[PAR_B2];
#pragma unroll
    for (int r = 0; r < 8; ++r) acc[r] = b2v;
#pragma unroll
    for (int l = 0; l < KGN; ++l) {
        __half2 hv[4];
        __half2 b1l = b1v[l];
#pragma unroll
        for (int rr = 0; rr < 4; ++rr) hv[rr] = b1l;
#pragma unroll
        for (int k = 0; k < KGN; ++k) {
            __half2 w = w1v[l * KGN + k];
#pragma unroll
            for (int rr = 0; rr < 4; ++rr)
                hv[rr] = __hfma2(psi[k][rr], w, hv[rr]);   // v_pk_fma_f16
        }
        float w2 = parv[PAR_W2 + l];
#pragma unroll
        for (int rr = 0; rr < 4; ++rr) {
            float2 hf = __half22float2(hv[rr]);
            float g0 = 0.5f * hf.x * (1.0f + erff(hf.x * 0.70710678118654752f));
            float g1 = 0.5f * hf.y * (1.0f + erff(hf.y * 0.70710678118654752f));
            acc[2 * rr]     = fmaf(g0, w2, acc[2 * rr]);
            acc[2 * rr + 1] = fmaf(g1, w2, acc[2 * rr + 1]);
        }
    }
#pragma unroll
    for (int r = 0; r < 8; ++r) out[r] += acc[r];
}

__global__ __launch_bounds__(128)
void gauss_flash_attn(const float* __restrict__ Qp, const float* __restrict__ Kp,
                      const float* __restrict__ Vp, const float* __restrict__ Dist,
                      const float* __restrict__ En, const int* __restrict__ Mask,
                      const float* __restrict__ muD, const float* __restrict__ sigD,
                      const float* __restrict__ bD,  const float* __restrict__ muE,
                      const float* __restrict__ sigE, const float* __restrict__ bE,
                      const float* __restrict__ W1,  const float* __restrict__ b1,
                      const float* __restrict__ W2,  const float* __restrict__ b2,
                      float* __restrict__ Out) {
    __shared__ __bf16  ldsK[32 * KP];        // K tile, [n][d]
    __shared__ __bf16  ldsV[64 * VP];        // V tile transposed, [d][k]
    __shared__ __bf16  ldsP[4 * 16 * PP];    // per-wave P scratch, [m][k]
    __shared__ float   par[PAR_N];
    __shared__ __half2 parW1h[KGN * KGN];    // W1 duplicated into both fp16 halves
    __shared__ __half2 parB1h[KGN];

    const int tid  = threadIdx.x;
    const int wave = tid >> 5;
    const int lane = tid & 31;
    const int half = lane >> 4;   // C/D layout: half 0 -> rows 0..7, half 1 -> rows 8..15
    const int ln   = lane & 15;
    const int hh   = blockIdx.y;
    const int row0 = blockIdx.x * 64 + wave * 16;

    // ---- stage RBF/MLP parameters ----
    if (tid < KGN) {
        int k = tid;
        float sd = sigD[k], se = sigE[k];
        par[PAR_AD + k] = bD[k] - muD[k];
        par[PAR_BD + k] = 1.0f / sd;
        par[PAR_CD + k] = 0.3989422804014327f / sd;   // 1/(sqrt(2pi)*sigma)
        par[PAR_AE + k] = bE[k] - muE[k];
        par[PAR_BE + k] = 1.0f / se;
        par[PAR_CE + k] = 0.3989422804014327f / se;
        par[PAR_W2 + k] = W2[k];
        parB1h[k] = __float2half2_rn(b1[k]);
    }
    if (tid < KGN * KGN) parW1h[tid] = __float2half2_rn(W1[tid]);
    if (tid == 0) par[PAR_B2] = b2[0];

    // ---- load this wave's 16 Q rows into bf16 A-layout, pre-scaled ----
    const float SCALE = 0.08838834764831845f;  // 1/sqrt(2*D) = 1/sqrt(128)
    v16bf qa[2];
    {
        const float* qrow = Qp + ((size_t)hh * SEQ + row0 + ln) * DH;
        const int kb = half * 8;
#pragma unroll
        for (int t = 0; t < 2; ++t) {
#pragma unroll
            for (int e = 0; e < 8; ++e) {
                qa[t][e]     = (__bf16)(qrow[t * 32 + kb + e]      * SCALE);
                qa[t][e + 8] = (__bf16)(qrow[t * 32 + kb + 16 + e] * SCALE);
            }
        }
    }

    v8f oacc[4];
#pragma unroll
    for (int s = 0; s < 4; ++s)
#pragma unroll
        for (int r = 0; r < 8; ++r) oacc[s][r] = 0.0f;

    float mrow[8], lrow[8];
#pragma unroll
    for (int r = 0; r < 8; ++r) { mrow[r] = -3.0e38f; lrow[r] = 0.0f; }

    const float* distH = Dist + (size_t)hh * SEQ * SEQ;
    const float* enH   = En   + (size_t)hh * SEQ * SEQ;
    const int*   mskH  = Mask + (size_t)hh * SEQ * SEQ;
    __bf16* myP = ldsP + wave * 16 * PP;

    for (int jt = 0; jt < SEQ / 32; ++jt) {
        const int col0 = jt * 32;
        __syncthreads();
        // ---- cooperative staging of K (as [n][d]) and V (transposed [d][k]) ----
        {
            const int n = tid & 31, c = tid >> 5;   // c in 0..3 -> 16-wide d-chunk
            const float* kr = Kp + ((size_t)hh * SEQ + col0 + n) * DH + c * 16;
#pragma unroll
            for (int e = 0; e < 16; e += 2) {
                union { __bf16 h2[2]; unsigned u; } pk;
                pk.h2[0] = (__bf16)kr[e];
                pk.h2[1] = (__bf16)kr[e + 1];
                *(unsigned*)&ldsK[n * KP + c * 16 + e] = pk.u;
            }
            const float* vr = Vp + ((size_t)hh * SEQ + col0 + n) * DH + c * 16;
#pragma unroll
            for (int e = 0; e < 16; ++e)
                ldsV[(c * 16 + e) * VP + n] = (__bf16)vr[e];
            if (jt + 1 < SEQ / 32) {
                __builtin_prefetch(Kp + ((size_t)hh * SEQ + col0 + 32 + n) * DH + c * 16, 0, 1);
                __builtin_prefetch(Vp + ((size_t)hh * SEQ + col0 + 32 + n) * DH + c * 16, 0, 1);
            }
        }
        __syncthreads();

        // ---- S = Q K^T : two 16x16 subtiles, accumulate over D=64 (2 x K=32) ----
        v8f sacc[2];
#pragma unroll
        for (int jj = 0; jj < 2; ++jj) {
            v8f c;
#pragma unroll
            for (int r = 0; r < 8; ++r) c[r] = 0.0f;
#pragma unroll
            for (int t = 0; t < 2; ++t) {
                // B-layout: lanes 0-15 K=0..7,16..23, lanes 16-31 K=8..15,24..31, N = ln
                const __bf16* bp = ldsK + (jj * 16 + ln) * KP + t * 32 + half * 8;
                v16bf bmat = mk16(*(const v8bf*)bp, *(const v8bf*)(bp + 16));
                c = __builtin_amdgcn_wmma_f32_16x16x32_bf16(
                        false, qa[t], false, bmat, (short)0, c, false, false);
            }
            sacc[jj] = c;
        }

        // ---- fused RBF-MLP biases (distance + energy) and mask ----
#pragma unroll
        for (int jj = 0; jj < 2; ++jj) {
            const int j = col0 + jj * 16 + ln;
            float xs[8], bias[8];
#pragma unroll
            for (int r = 0; r < 8; ++r) bias[r] = 0.0f;
#pragma unroll
            for (int r = 0; r < 8; ++r)
                xs[r] = distH[(size_t)(row0 + half * 8 + r) * SEQ + j];
            bias8(xs, bias, par, parW1h, parB1h, PAR_AD);
#pragma unroll
            for (int r = 0; r < 8; ++r)
                xs[r] = enH[(size_t)(row0 + half * 8 + r) * SEQ + j];
            bias8(xs, bias, par, parW1h, parB1h, PAR_AE);
#pragma unroll
            for (int r = 0; r < 8; ++r) {
                float sc = sacc[jj][r] + bias[r];
                int mv = mskH[(size_t)(row0 + half * 8 + r) * SEQ + j];
                sacc[jj][r] = (mv == 0) ? -1.0e9f : sc;
            }
        }

        // ---- online softmax (rows live in 16-lane halves) ----
#pragma unroll
        for (int r = 0; r < 8; ++r) {
            float mx = fmaxf(sacc[0][r], sacc[1][r]);
#pragma unroll
            for (int off = 8; off >= 1; off >>= 1)
                mx = fmaxf(mx, __shfl_xor(mx, off, 32));
            float mnew = fmaxf(mrow[r], mx);
            float corr = __expf(mrow[r] - mnew);
            mrow[r] = mnew;
            lrow[r] *= corr;
#pragma unroll
            for (int s = 0; s < 4; ++s) oacc[s][r] *= corr;
            float p0 = __expf(sacc[0][r] - mnew);
            float p1 = __expf(sacc[1][r] - mnew);
            sacc[0][r] = p0; sacc[1][r] = p1;
            float rs = p0 + p1;
#pragma unroll
            for (int off = 8; off >= 1; off >>= 1)
                rs += __shfl_xor(rs, off, 32);
            lrow[r] += rs;
        }

        // ---- C-layout P -> LDS -> A-layout bf16 (in-wave DS ordering) ----
#pragma unroll
        for (int jj = 0; jj < 2; ++jj)
#pragma unroll
            for (int r = 0; r < 8; ++r)
                myP[(half * 8 + r) * PP + jj * 16 + ln] = (__bf16)sacc[jj][r];
        asm volatile("s_wait_dscnt 0" ::: "memory");
        v16bf pa;
        {
            const __bf16* pp = myP + ln * PP + half * 8;
            pa = mk16(*(const v8bf*)pp, *(const v8bf*)(pp + 16));
        }

        // ---- O += P V : 4 subtiles along D ----
#pragma unroll
        for (int s = 0; s < 4; ++s) {
            const __bf16* vp = ldsV + (s * 16 + ln) * VP + half * 8;
            v16bf bmat = mk16(*(const v8bf*)vp, *(const v8bf*)(vp + 16));
            oacc[s] = __builtin_amdgcn_wmma_f32_16x16x32_bf16(
                          false, pa, false, bmat, (short)0, oacc[s], false, false);
        }
    }

    // ---- normalize and write out ----
    float inv[8];
#pragma unroll
    for (int r = 0; r < 8; ++r) inv[r] = 1.0f / lrow[r];
#pragma unroll
    for (int s = 0; s < 4; ++s)
#pragma unroll
        for (int r = 0; r < 8; ++r)
            Out[((size_t)hh * SEQ + row0 + half * 8 + r) * DH + s * 16 + ln] =
                oacc[s][r] * inv[r];
}

extern "C" void kernel_launch(void* const* d_in, const int* in_sizes, int n_in,
                              void* d_out, int out_size, void* d_ws, size_t ws_size,
                              hipStream_t stream) {
    (void)in_sizes; (void)n_in; (void)out_size; (void)d_ws; (void)ws_size;
    const float* Q    = (const float*)d_in[0];
    const float* K    = (const float*)d_in[1];
    const float* V    = (const float*)d_in[2];
    const float* Dist = (const float*)d_in[3];
    const float* En   = (const float*)d_in[4];
    const int*   Msk  = (const int*)d_in[5];
    const float* muD  = (const float*)d_in[6];
    const float* sigD = (const float*)d_in[7];
    const float* bD   = (const float*)d_in[8];
    const float* muE  = (const float*)d_in[9];
    const float* sigE = (const float*)d_in[10];
    const float* bE   = (const float*)d_in[11];
    const float* W1   = (const float*)d_in[12];
    const float* b1   = (const float*)d_in[13];
    const float* W2   = (const float*)d_in[14];
    const float* b2   = (const float*)d_in[15];
    float* Out = (float*)d_out;

    dim3 grid(SEQ / 64, BH);   // 16 row-blocks x 8 heads
    dim3 block(128);           // 4 waves, one 16-row tile each
    gauss_flash_attn<<<grid, block, 0, stream>>>(Q, K, V, Dist, En, Msk,
                                                 muD, sigD, bD, muE, sigE, bE,
                                                 W1, b1, W2, b2, Out);
}
